// MultiHeadAttentionEnc_50586124812558
// MI455X (gfx1250) — compile-verified, matrix operand order checked by reading, not run on hardware
//
#include <hip/hip_runtime.h>
#include <hip/hip_bf16.h>

// ---------------------------------------------------------------------------
// Types for CDNA5 WMMA (wave32, 16x16x32 bf16 -> f32)
// ---------------------------------------------------------------------------
typedef __attribute__((ext_vector_type(16))) __bf16 v16bf;
typedef __attribute__((ext_vector_type(8)))  __bf16 v8bf;
typedef __attribute__((ext_vector_type(8)))  float  v8f;
typedef int v4i_vec __attribute__((vector_size(4 * sizeof(int))));

#if __has_builtin(__builtin_amdgcn_global_load_async_to_lds_b128)
#define HAS_ASYNC_LDS 1
#else
#define HAS_ASYNC_LDS 0
#endif

#if HAS_ASYNC_LDS
static __device__ __forceinline__ void async_copy16(const __bf16* g, __bf16* l) {
    __builtin_amdgcn_global_load_async_to_lds_b128(
        (__attribute__((address_space(1))) v4i_vec*)g,
        (__attribute__((address_space(3))) v4i_vec*)l, 0, 0);
}
static __device__ __forceinline__ void wait_async0() {
#if __has_builtin(__builtin_amdgcn_s_wait_asynccnt)
    __builtin_amdgcn_s_wait_asynccnt(0);
#else
    asm volatile("s_wait_asynccnt 0" ::: "memory");
#endif
}
#endif

static __device__ __forceinline__ v16bf mk16(v8bf lo, v8bf hi) {
    v16bf r;
#pragma unroll
    for (int i = 0; i < 8; ++i) { r[i] = lo[i]; r[8 + i] = hi[i]; }
    return r;
}
static __device__ __forceinline__ v8f vzero8() {
    v8f z;
#pragma unroll
    for (int i = 0; i < 8; ++i) z[i] = 0.0f;
    return z;
}
// Butterfly reductions over the 16 lanes of a wave32 half (xor masks < 16)
static __device__ __forceinline__ float hmax16(float v) {
#pragma unroll
    for (int m = 1; m <= 8; m <<= 1) v = fmaxf(v, __shfl_xor(v, m, 32));
    return v;
}
static __device__ __forceinline__ float hsum16(float v) {
#pragma unroll
    for (int m = 1; m <= 8; m <<= 1) v += __shfl_xor(v, m, 32);
    return v;
}

static constexpr int E  = 1024;
static constexpr int Hh = 8;
static constexpr int Dd = 128;
static constexpr int Mm = 1024;
static constexpr int Bb = 32;
static constexpr int Ntok = Bb * Mm;  // 32768

// ---------------------------------------------------------------------------
// Kernel: fp32 -> bf16 weight conversion
// ---------------------------------------------------------------------------
__global__ void cvt_f32_bf16(const float* __restrict__ src,
                             __bf16* __restrict__ dst, int n) {
    int i = blockIdx.x * blockDim.x + threadIdx.x;
    if (i < n) dst[i] = (__bf16)src[i];
}

// ---------------------------------------------------------------------------
// Kernel: fused GEMM (X @ W^T + b) [+ CELU + GroupNorm(H) + affine]
// Block: 8 waves = 256 thr; tile = 32 tokens x 128 channels (one GN group).
// Each wave: 2 WMMAs per K-step sharing one W fragment.
// grid = (Ntok/32, E/128)
// GN=true : write bf16 to out_qkv laid out [B, H, M, D]
// GN=false: write f32  to out_f32 laid out [Ntok, E]
// ---------------------------------------------------------------------------
template <bool GN, typename XT>
__global__ __launch_bounds__(256)
void proj_kernel(const XT* __restrict__ X, const __bf16* __restrict__ Wb,
                 const float* __restrict__ bias, const float* __restrict__ gamma,
                 const float* __restrict__ beta, __bf16* __restrict__ out_qkv,
                 float* __restrict__ out_f32) {
    __shared__ __align__(16) __bf16 Xt[32][32];
    __shared__ float redS[8][32];
    __shared__ float redQ[8][32];
    __shared__ float muS[32];
    __shared__ float rsS[32];

    const int tid  = threadIdx.x;
    const int w    = tid >> 5;
    const int lane = tid & 31;
    const int half = lane >> 4;
    const int ln   = lane & 15;
    const int klo  = half * 8;

    const int n0 = blockIdx.x * 32;                  // first token of tile
    const int cg = blockIdx.y * 128 + w * 16 + ln;   // this lane's out channel

    v8f acc0 = vzero8();   // tokens n0 + [0,16)
    v8f acc1 = vzero8();   // tokens n0 + [16,32)

    const int xr = tid >> 3;         // 0..31 (row for staging, 4 elems each)
    const int xc = (tid & 7) * 4;    // col group

    for (int kt = 0; kt < 32; ++kt) {
        __syncthreads();
        // stage 32x32 X tile as bf16 (4 elements per thread)
#pragma unroll
        for (int j = 0; j < 4; ++j)
            Xt[xr][xc + j] = (__bf16)(float)X[(size_t)(n0 + xr) * E + kt * 32 + xc + j];
        if (kt < 31)  // prefetch next X chunk (global_prefetch_b8)
            __builtin_prefetch(&X[(size_t)(n0 + xr) * E + (kt + 1) * 32 + xc], 0, 1);
        __syncthreads();

        // Two A fragments (rows 0-15 and 16-31), one shared B fragment
        v16bf af0 = mk16(*(const v8bf*)&Xt[ln][klo],
                         *(const v8bf*)&Xt[ln][16 + klo]);
        v16bf af1 = mk16(*(const v8bf*)&Xt[16 + ln][klo],
                         *(const v8bf*)&Xt[16 + ln][16 + klo]);
        const __bf16* wr = Wb + (size_t)cg * E + kt * 32;
        v16bf bf = mk16(*(const v8bf*)(wr + klo),
                        *(const v8bf*)(wr + 16 + klo));
        if (kt < 31) __builtin_prefetch(wr + 32, 0, 1);

        acc0 = __builtin_amdgcn_wmma_f32_16x16x32_bf16(
            false, af0, false, bf, (short)0, acc0, false, false);
        acc1 = __builtin_amdgcn_wmma_f32_16x16x32_bf16(
            false, af1, false, bf, (short)0, acc1, false, false);
    }

    float y0[8], y1[8];
    const float bv = bias[cg];
#pragma unroll
    for (int r = 0; r < 8; ++r) { y0[r] = acc0[r] + bv; y1[r] = acc1[r] + bv; }

    if constexpr (GN) {
        // CELU(alpha=1.3)
#pragma unroll
        for (int r = 0; r < 8; ++r) {
            y0[r] = y0[r] > 0.0f ? y0[r] : 1.3f * expm1f(y0[r] * (1.0f / 1.3f));
            y1[r] = y1[r] > 0.0f ? y1[r] : 1.3f * expm1f(y1[r] * (1.0f / 1.3f));
        }

        // per-token partial sums over this wave's 16 channels
#pragma unroll
        for (int r = 0; r < 8; ++r) {
            float s0 = hsum16(y0[r]);
            float q0 = hsum16(y0[r] * y0[r]);
            float s1 = hsum16(y1[r]);
            float q1 = hsum16(y1[r] * y1[r]);
            if (ln == 0) {
                redS[w][r + 8 * half]      = s0;
                redQ[w][r + 8 * half]      = q0;
                redS[w][16 + r + 8 * half] = s1;
                redQ[w][16 + r + 8 * half] = q1;
            }
        }
        __syncthreads();
        if (tid < 32) {
            float s = 0.0f, q = 0.0f;
#pragma unroll
            for (int ww = 0; ww < 8; ++ww) { s += redS[ww][tid]; q += redQ[ww][tid]; }
            float mu  = s * (1.0f / 128.0f);
            float var = q * (1.0f / 128.0f) - mu * mu;
            muS[tid] = mu;
            rsS[tid] = rsqrtf(var + 1e-5f);
        }
        __syncthreads();

        const float g  = gamma[cg];
        const float be = beta[cg];
        const int h = cg >> 7;      // head
        const int d = cg & 127;     // dim within head
#pragma unroll
        for (int r = 0; r < 8; ++r) {
            int t0 = r + 8 * half;
            int t1 = 16 + t0;
            float z0 = (y0[r] - muS[t0]) * rsS[t0] * g + be;
            float z1 = (y1[r] - muS[t1]) * rsS[t1] * g + be;
            int na = n0 + t0, nb = n0 + t1;
            int bia = na >> 10, ma = na & 1023;   // M = 1024
            int bib = nb >> 10, mb = nb & 1023;
            out_qkv[(((size_t)bia * Hh + h) * Mm + ma) * Dd + d] = (__bf16)z0;
            out_qkv[(((size_t)bib * Hh + h) * Mm + mb) * Dd + d] = (__bf16)z1;
        }
    } else {
#pragma unroll
        for (int r = 0; r < 8; ++r) {
            int na = n0 + r + 8 * half;
            int nb = na + 16;
            out_f32[(size_t)na * E + cg] = y0[r];
            out_f32[(size_t)nb * E + cg] = y1[r];
        }
    }
}

// ---------------------------------------------------------------------------
// Kernel: flash-style attention, one (b, h, 128-query block) per workgroup.
// 8 waves x 16 query rows; streams 32-key tiles through LDS.
// K tile staged with async global->LDS (ASYNCcnt) when available.
// q/k/v: bf16 [B, H, M, D].  Output: bf16 [Ntok, E] (= transpose-merge).
// ---------------------------------------------------------------------------
__global__ __launch_bounds__(256)
void attn_kernel(const __bf16* __restrict__ qbuf, const __bf16* __restrict__ kbuf,
                 const __bf16* __restrict__ vbuf, const int* __restrict__ mask,
                 __bf16* __restrict__ attout) {
    __shared__ __align__(16) __bf16 Kt[32][128];     // keys as-is
    __shared__ __align__(16) __bf16 VT[128][32];     // V transposed: [d][key]
    __shared__ __align__(16) __bf16 Pst[8][16][32];  // per-wave P staging

    const int tid  = threadIdx.x;
    const int w    = tid >> 5;
    const int lane = tid & 31;
    const int half = lane >> 4;
    const int ln   = lane & 15;
    const int klo  = half * 8;

    const int blk = blockIdx.x;
    const int qb  = blk & 7;
    const int h   = (blk >> 3) & 7;
    const int b   = blk >> 6;
    const size_t bh    = ((size_t)b * Hh + h) * Mm;  // row base in [B*H*M, D]
    const int    qrow0 = qb * 128 + w * 16;

    // hold all Q fragments for this wave's 16 rows (D=128 -> 4 K-steps)
    v16bf aq[4];
#pragma unroll
    for (int dt = 0; dt < 4; ++dt) {
        const __bf16* qp = qbuf + (bh + qrow0 + ln) * Dd + dt * 32;
        aq[dt] = mk16(*(const v8bf*)(qp + klo), *(const v8bf*)(qp + 16 + klo));
    }

    v8f o[8];
#pragma unroll
    for (int j = 0; j < 8; ++j) o[j] = vzero8();
    float mrow[8], lsum[8];
#pragma unroll
    for (int r = 0; r < 8; ++r) { mrow[r] = -3.0e38f; lsum[r] = 0.0f; }

    const float scal = 0.08838834764831845f;  // 1/sqrt(128)

    for (int kt = 0; kt < 32; ++kt) {
        __syncthreads();
        // ---- stage K tile (32x128) as 16B chunks: async global->LDS if avail
#pragma unroll
        for (int ci = tid; ci < 512; ci += 256) {
            int e0 = ci * 8;
            int r  = e0 >> 7, c = e0 & 127;
            const __bf16* gp = kbuf + (bh + kt * 32 + r) * Dd + c;
#if HAS_ASYNC_LDS
            async_copy16(gp, &Kt[r][c]);
#else
            *(v8bf*)&Kt[r][c] = *(const v8bf*)gp;
#endif
            if (kt < 31)
                __builtin_prefetch(kbuf + (bh + (kt + 1) * 32 + r) * Dd + c, 0, 1);
        }
        // ---- stage V tile transposed: b128 chunk loads + scattered b16 stores
#pragma unroll
        for (int ci = tid; ci < 512; ci += 256) {
            int e0 = ci * 8;
            int key = e0 >> 7, c = e0 & 127;
            v8bf chunk = *(const v8bf*)(vbuf + (bh + kt * 32 + key) * Dd + c);
#pragma unroll
            for (int j = 0; j < 8; ++j) VT[c + j][key] = chunk[j];
            if (kt < 31)
                __builtin_prefetch(vbuf + (bh + (kt + 1) * 32 + key) * Dd + c, 0, 1);
        }
#if HAS_ASYNC_LDS
        wait_async0();
#endif
        __syncthreads();

        // ---- S = q @ k^T for 32 keys: load ALL 8 B fragments first
        v16bf kb[8];
#pragma unroll
        for (int dt = 0; dt < 4; ++dt) {
            const __bf16* kr0 = &Kt[ln][dt * 32];
            const __bf16* kr1 = &Kt[ln + 16][dt * 32];
            kb[2 * dt]     = mk16(*(const v8bf*)(kr0 + klo), *(const v8bf*)(kr0 + 16 + klo));
            kb[2 * dt + 1] = mk16(*(const v8bf*)(kr1 + klo), *(const v8bf*)(kr1 + 16 + klo));
        }
        v8f s0 = vzero8(), s1 = vzero8();
#pragma unroll
        for (int dt = 0; dt < 4; ++dt) {
            s0 = __builtin_amdgcn_wmma_f32_16x16x32_bf16(
                false, aq[dt], false, kb[2 * dt], (short)0, s0, false, false);
            s1 = __builtin_amdgcn_wmma_f32_16x16x32_bf16(
                false, aq[dt], false, kb[2 * dt + 1], (short)0, s1, false, false);
        }

        // ---- scale + mask (column-wise)
        const int key0 = kt * 32 + ln;
        const bool mz0 = (mask[b * Mm + key0] == 0);
        const bool mz1 = (mask[b * Mm + key0 + 16] == 0);
#pragma unroll
        for (int r = 0; r < 8; ++r) {
            float a0 = s0[r] * scal; if (mz0) a0 = -1e9f; s0[r] = a0;
            float a1 = s1[r] * scal; if (mz1) a1 = -1e9f; s1[r] = a1;
        }

        // ---- online softmax per row; stash exp-weights to per-wave LDS
#pragma unroll
        for (int r = 0; r < 8; ++r) {
            float tmax  = hmax16(fmaxf(s0[r], s1[r]));
            float mnew  = fmaxf(mrow[r], tmax);
            float alpha = __expf(mrow[r] - mnew);
            float p0 = __expf(s0[r] - mnew);
            float p1 = __expf(s1[r] - mnew);
            float rs = hsum16(p0 + p1);
            lsum[r] = lsum[r] * alpha + rs;
            mrow[r] = mnew;
#pragma unroll
            for (int j = 0; j < 8; ++j) o[j][r] *= alpha;
            Pst[w][r + 8 * half][ln]      = (__bf16)p0;
            Pst[w][r + 8 * half][ln + 16] = (__bf16)p1;
        }

        // ---- reload P in A-fragment layout (wave-private LDS region)
        const __bf16* pr = &Pst[w][ln][0];
        v16bf pf = mk16(*(const v8bf*)(pr + klo), *(const v8bf*)(pr + 16 + klo));

        // ---- O += P @ V: load ALL 8 V fragments first, then 8 WMMAs
        v16bf vfz[8];
#pragma unroll
        for (int j = 0; j < 8; ++j) {
            const __bf16* vr = &VT[j * 16 + ln][0];
            vfz[j] = mk16(*(const v8bf*)(vr + klo), *(const v8bf*)(vr + 16 + klo));
        }
#pragma unroll
        for (int j = 0; j < 8; ++j) {
            o[j] = __builtin_amdgcn_wmma_f32_16x16x32_bf16(
                false, pf, false, vfz[j], (short)0, o[j], false, false);
        }
    }

    // ---- normalize and store as [Ntok, E] bf16 (transpose(0,2,1,3).reshape)
    float rinv[8];
#pragma unroll
    for (int r = 0; r < 8; ++r) rinv[r] = 1.0f / lsum[r];
#pragma unroll
    for (int j = 0; j < 8; ++j) {
#pragma unroll
        for (int r = 0; r < 8; ++r) {
            int mg = qrow0 + r + 8 * half;
            size_t n = (size_t)b * Mm + mg;
            attout[n * E + h * 128 + j * 16 + ln] = (__bf16)(o[j][r] * rinv[r]);
        }
    }
}

// ---------------------------------------------------------------------------
// Host launcher
// ---------------------------------------------------------------------------
extern "C" void kernel_launch(void* const* d_in, const int* in_sizes, int n_in,
                              void* d_out, int out_size, void* d_ws, size_t ws_size,
                              hipStream_t stream) {
    const float* query  = (const float*)d_in[0];
    const float* key    = (const float*)d_in[1];
    const int*   mask   = (const int*)d_in[2];
    /* value1 (d_in[3]) intentionally unused */
    const float* value2 = (const float*)d_in[4];
    const float* Wq = (const float*)d_in[5];
    const float* bq = (const float*)d_in[6];
    const float* gq = (const float*)d_in[7];
    const float* betaq = (const float*)d_in[8];
    const float* Wk = (const float*)d_in[9];
    const float* bk = (const float*)d_in[10];
    const float* gk = (const float*)d_in[11];
    const float* betak = (const float*)d_in[12];
    const float* Wv = (const float*)d_in[13];
    const float* bv = (const float*)d_in[14];
    const float* gv = (const float*)d_in[15];
    const float* betav = (const float*)d_in[16];
    const float* Wm = (const float*)d_in[17];
    const float* bm = (const float*)d_in[18];

    float* out = (float*)d_out;

    // workspace layout
    const size_t MB = 1u << 20;
    char* ws = (char*)d_ws;
    __bf16* Wqb = (__bf16*)(ws + 0 * MB);
    __bf16* Wkb = (__bf16*)(ws + 2 * MB);
    __bf16* Wvb = (__bf16*)(ws + 4 * MB);
    __bf16* Wmb = (__bf16*)(ws + 6 * MB);
    __bf16* qbuf = (__bf16*)(ws + 8 * MB);    // [B,H,M,D]
    __bf16* kbuf = (__bf16*)(ws + 72 * MB);
    __bf16* vbuf = (__bf16*)(ws + 136 * MB);
    __bf16* attb = (__bf16*)(ws + 200 * MB);  // [Ntok, E]

    // weights -> bf16
    const int nw = E * E;
    const int cvtBlocks = (nw + 255) / 256;
    cvt_f32_bf16<<<cvtBlocks, 256, 0, stream>>>(Wq, Wqb, nw);
    cvt_f32_bf16<<<cvtBlocks, 256, 0, stream>>>(Wk, Wkb, nw);
    cvt_f32_bf16<<<cvtBlocks, 256, 0, stream>>>(Wv, Wvb, nw);
    cvt_f32_bf16<<<cvtBlocks, 256, 0, stream>>>(Wm, Wmb, nw);

    dim3 pgrid(Ntok / 32, E / 128);
    proj_kernel<true, float><<<pgrid, 256, 0, stream>>>(
        query, Wqb, bq, gq, betaq, qbuf, nullptr);
    proj_kernel<true, float><<<pgrid, 256, 0, stream>>>(
        key, Wkb, bk, gk, betak, kbuf, nullptr);
    proj_kernel<true, float><<<pgrid, 256, 0, stream>>>(
        value2, Wvb, bv, gv, betav, vbuf, nullptr);

    attn_kernel<<<Bb * Hh * (Mm / 128), 256, 0, stream>>>(
        qbuf, kbuf, vbuf, mask, attb);

    proj_kernel<false, __bf16><<<pgrid, 256, 0, stream>>>(
        attb, Wmb, bm, nullptr, nullptr, nullptr, out);
}
